// MultiScaleFusionRoIAlign_77017353552546
// MI455X (gfx1250) — matrix operand order repaired and checked by reading, not compiled, not run on hardware
//
#include <hip/hip_runtime.h>
#include <hip/hip_bf16.h>
#include <cstddef>

// ---------------------------------------------------------------------------
// Fused multi-scale RoI-Align for MI455X (gfx1250, wave32).
//
// No upsampled-pyramid materialization (saves ~334MB HBM traffic; the 146MB
// pyramid is L2-resident on the 192MB L2). Separable bilinear interpolation
// cast as two K=32 f16 WMMA GEMMs per (roi, level, 16-channel tile):
//   GEMM-A: Ty[j][oh][c] = sum_y Wy[oh][y] * G[y][j][c]    (32 wmma, K=32)
//   GEMM-B: Out[ow][c]   = sum_j Wx[ow][j] * Ty[j][oh][c]  ( 7 wmma, K=32)
// 14 samples x 2 taps = 28 taps per axis, padded to K=32 with zero weights.
// LDS layouts are K-contiguous per lane so B fragments load as ds_load_b128.
// ---------------------------------------------------------------------------

typedef _Float16 half_t;
typedef __attribute__((ext_vector_type(2)))  _Float16 v2h;
typedef __attribute__((ext_vector_type(8)))  _Float16 v8h;
typedef __attribute__((ext_vector_type(16))) _Float16 v16h;
typedef __attribute__((ext_vector_type(8)))  float    v8f;

#define RA_B   2
#define RA_N   64
#define RA_C   256      // channels per level
#define RA_OH  7
#define RA_SR  2
#define RA_H0  200
#define RA_W0  272
#define RA_CT  16       // channel tile (WMMA N)
#define RA_NT  (RA_C / RA_CT)

// A fragment (16x32 f16): lanes 0-15 hold M=lane, K={ka..ka+15}
// with ka=0; lanes 16-31 the same rows with ka=16.
static __device__ inline v16h load_a_frag(const half_t* W, int lane) {
    const int row = lane & 15;
    const int ka  = (lane < 16) ? 0 : 16;
    const v8h lo = *(const v8h*)(W + row * 32 + ka);
    const v8h hi = *(const v8h*)(W + row * 32 + 8 + ka);
    return __builtin_shufflevector(lo, hi, 0, 1, 2, 3, 4, 5, 6, 7,
                                           8, 9, 10, 11, 12, 13, 14, 15);
}

__global__ __launch_bounds__(32)
void roi_align_wmma_kernel(const float* __restrict__ x0,
                           const float* __restrict__ x1,
                           const float* __restrict__ x2,
                           const float* __restrict__ boxes,
                           const float* __restrict__ ratios,
                           const float* __restrict__ offsets,
                           const int*   __restrict__ shapes,
                           float* __restrict__ out)
{
    // ---- block decode: (b, roi, level, channel tile) ----
    const int bi   = blockIdx.x;
    const int tile = bi & (RA_NT - 1);
    const int lvl  = (bi >> 4) % 3;
    const int roi  = (bi / (RA_NT * 3)) & (RA_N - 1);
    const int b    = bi / (RA_NT * 3 * RA_N);
    const int lane = threadIdx.x;

    const float* lp; int Hl, Wl;
    if (lvl == 0)      { lp = x0; Hl = 200; Wl = 272; }
    else if (lvl == 1) { lp = x1; Hl = 100; Wl = 136; }
    else               { lp = x2; Hl = 50;  Wl = 68;  }
    const int cbase = tile * RA_CT;

    // ---- LDS (K-contiguous layouts for vector fragment loads) ----
    __shared__ int    s_yidx[32], s_xidx[32];
    __shared__ float  s_wy[2][16], s_wx[2][16];              // [lo/hi][sample]
    __shared__ __align__(32) half_t s_Wy[16 * 32];           // [oh][ytap]
    __shared__ __align__(32) half_t s_Wx[16 * 32];           // [ow][xtap]
    __shared__ __align__(32) half_t s_G[32 * RA_CT * 32];    // [j][c][y] 32KB
    __shared__ __align__(32) half_t s_U[16 * RA_CT * 32];    // [oh][c][j] 16KB

    // ---- per-ROI geometry (all lanes compute; lanes 0..13 publish) ----
    {
        const float rH   = ratios[b * 2 + 0], rW = ratios[b * 2 + 1];
        const float top  = offsets[b * 2 + 0], left = offsets[b * 2 + 1];
        const float* bp  = boxes + ((size_t)b * RA_N + roi) * 4;
        const float rx0  = bp[0] * rW + left, ry0 = bp[1] * rH + top;
        const float rx1  = bp[2] * rW + left, ry1 = bp[3] * rH + top;
        const float fullh = (float)shapes[0] * ratios[0] + 2.f * offsets[0];
        const float fullw = (float)shapes[1] * ratios[1] + 2.f * offsets[1];
        const float sw = (float)RA_W0 / fullw, sh = (float)RA_H0 / fullh;
        const float bx0 = rx0 * sw, by0 = ry0 * sh;
        const float rw = fmaxf(rx1 * sw - bx0, 1.f);
        const float rh = fmaxf(ry1 * sh - by0, 1.f);
        const float binh = rh / RA_OH, binw = rw / RA_OH;

        if (lane < RA_OH * RA_SR) {                  // 14 samples per axis
            const float fs = (float)lane + 0.5f;
            const float ys = by0 + fs * binh * 0.5f; // fine-grid position
            const float xs = bx0 + fs * binw * 0.5f;
            const bool vy = (ys >= -1.f) && (ys <= (float)RA_H0);
            const bool vx = (xs >= -1.f) && (xs <= (float)RA_W0);
            // fold upsample into a direct bilinear sample at level coords
            float fy = (ys + 0.5f) * ((float)Hl / (float)RA_H0) - 0.5f;
            float fx = (xs + 0.5f) * ((float)Wl / (float)RA_W0) - 0.5f;
            fy = fminf(fmaxf(fy, 0.f), (float)(Hl - 1));
            fx = fminf(fmaxf(fx, 0.f), (float)(Wl - 1));
            const float fly = floorf(fy), flx = floorf(fx);
            const int yl = (int)fly, xl = (int)flx;
            const float qy = fy - fly, qx = fx - flx;
            s_yidx[2 * lane]     = yl;
            s_yidx[2 * lane + 1] = min(yl + 1, Hl - 1);
            s_xidx[2 * lane]     = xl;
            s_xidx[2 * lane + 1] = min(xl + 1, Wl - 1);
            s_wy[0][lane] = vy ? (1.f - qy) : 0.f;
            s_wy[1][lane] = vy ? qy         : 0.f;
            s_wx[0][lane] = vx ? (1.f - qx) : 0.f;
            s_wx[1][lane] = vx ? qx         : 0.f;
        }
        if (lane >= 28) { s_yidx[lane] = 0; s_xidx[lane] = 0; }  // K pad
    }
    __syncthreads();

    // ---- build Wy / Wx: [M=16 bins (7 live)][K=32 taps]; 0.5 = sample avg --
    {
        const int j = lane;              // tap column
        const int s = j >> 1, t = j & 1; // sample, lo/hi tap
        #pragma unroll
        for (int r = 0; r < 16; ++r) {
            const bool live = (j < 28) && (r == (s >> 1));
            s_Wy[r * 32 + j] = (half_t)(live ? 0.5f * s_wy[t][s] : 0.f);
            s_Wx[r * 32 + j] = (half_t)(live ? 0.5f * s_wx[t][s] : 0.f);
        }
    }
    __syncthreads();

    // ---- gather G[j][c][y] from the L2-resident level; half2 packed stores -
    {
        const int c  = lane & (RA_CT - 1);
        const int yh = lane >> 4;                     // y half: 0..15 / 16..31
        const size_t coff =
            (size_t)((b * RA_C) + cbase + c) * (size_t)(Hl * Wl);
        int ry[16];                                   // this lane's row bases
        #pragma unroll
        for (int y2 = 0; y2 < 16; ++y2)
            ry[y2] = s_yidx[(yh << 4) + y2] * Wl;
        const int ybase = yh << 4;
        for (int j = 0; j < 32; ++j) {
            const int xi = s_xidx[j];
            #pragma unroll
            for (int p = 0; p < 8; ++p) {             // y pairs
                const float f0 = lp[coff + (size_t)(ry[2 * p]     + xi)];
                const float f1 = lp[coff + (size_t)(ry[2 * p + 1] + xi)];
                v2h pk; pk[0] = (half_t)f0; pk[1] = (half_t)f1;
                *(v2h*)&s_G[(j << 9) + (c << 5) + ybase + (p << 1)] = pk;
            }
        }
    }
    __syncthreads();

    // ---- fragment constants ----
    const int fr_col = lane & 15;                 // A row / B,D column (N)
    const int kb     = (lane < 16) ? 0 : 16;      // B K base

    const v16h a_wy = load_a_frag(s_Wy, lane);
    const v16h a_wx = load_a_frag(s_Wx, lane);

    // ---- GEMM-A: contract y-taps. One K=32 WMMA per x-tap column j. ----
    for (int j = 0; j < 32; ++j) {
        // B[k][n] = G[j][n][k] : contiguous 16 halfs -> 2x ds_load_b128
        const v16h bfrag = *(const v16h*)&s_G[(j << 9) + (fr_col << 5) + kb];
        v8f acc = {};
        acc = __builtin_amdgcn_wmma_f32_16x16x32_f16(
            false, a_wy, false, bfrag, (short)0, acc, false, false);
        #pragma unroll
        for (int r = 0; r < 8; ++r) {              // U[oh][c][j] (f16)
            const int row = (lane < 16) ? r : (8 + r);
            s_U[(row << 9) + (fr_col << 5) + j] = (half_t)acc[r];
        }
    }
    __syncthreads();

    // ---- GEMM-B: contract x-taps; one K=32 WMMA per output row oh ----
    const size_t obase =
        (((size_t)b * RA_N + roi) * (3 * RA_C) + (size_t)lvl * RA_C + cbase)
        * (RA_OH * RA_OH);
    for (int oh = 0; oh < RA_OH; ++oh) {
        // B[k][n] = U[oh][n][k] : contiguous 16 halfs -> 2x ds_load_b128
        const v16h bfrag = *(const v16h*)&s_U[(oh << 9) + (fr_col << 5) + kb];
        v8f acc = {};
        acc = __builtin_amdgcn_wmma_f32_16x16x32_f16(
            false, a_wx, false, bfrag, (short)0, acc, false, false);
        #pragma unroll
        for (int r = 0; r < 8; ++r) {
            const int ow = (lane < 16) ? r : (8 + r);   // D row = M = ow
            if (ow < RA_OH) {                           // rows 7..15 are pad
                out[obase + (size_t)fr_col * (RA_OH * RA_OH)
                          + (size_t)oh * RA_OH + ow] = acc[r];
            }
        }
    }
}

extern "C" void kernel_launch(void* const* d_in, const int* in_sizes, int n_in,
                              void* d_out, int out_size, void* d_ws, size_t ws_size,
                              hipStream_t stream) {
    (void)in_sizes; (void)n_in; (void)d_ws; (void)ws_size; (void)out_size;
    const float* x0      = (const float*)d_in[0];
    const float* x1      = (const float*)d_in[1];
    const float* x2      = (const float*)d_in[2];
    const float* boxes   = (const float*)d_in[3];
    const float* ratios  = (const float*)d_in[4];
    const float* offsets = (const float*)d_in[5];
    const int*   shapes  = (const int*)d_in[6];
    float* out = (float*)d_out;

    // grid = B * N * 3 levels * 16 channel-tiles; one wave32 per block
    dim3 grid(RA_B * RA_N * 3 * RA_NT);
    dim3 block(32);
    roi_align_wmma_kernel<<<grid, block, 0, stream>>>(
        x0, x1, x2, boxes, ratios, offsets, shapes, out);
}
